// Model_16286515987119
// MI455X (gfx1250) — compile-verified, matrix operand order checked by reading, not run on hardware
//
#include <hip/hip_runtime.h>
#include <hip/hip_bf16.h>

typedef __attribute__((ext_vector_type(16))) _Float16     v16h;
typedef __attribute__((ext_vector_type(8)))  float        v8f;
typedef __attribute__((ext_vector_type(4)))  unsigned int v4u;
typedef __attribute__((ext_vector_type(8)))  int          v8i;
typedef __attribute__((ext_vector_type(4)))  int          v4i;

// ---------------------------------------------------------------------------
// f32 -> f16 weight conversion (one-time into workspace)
// ---------------------------------------------------------------------------
__global__ void f32_to_f16_kernel(const float* __restrict__ s,
                                  unsigned short* __restrict__ d, int nel) {
    int i = blockIdx.x * 256 + threadIdx.x;
    if (i < nel) {
        _Float16 h = (_Float16)s[i];
        d[i] = __builtin_bit_cast(unsigned short, h);
    }
}

// ---------------------------------------------------------------------------
// Fused capsule conv: im2col -> votes via WMMA (f16 in, f32 acc, votes in LDS)
// -> 3 routing iterations (softmax over i) -> (optional squash) -> out.
//
// Block = 128 threads (4 waves). Each block: 16 output pixels x 1 output
// capsule n. votes[16][I][e] (f32) live entirely in LDS; weight slice
// w[n][I][d][e] (f16) staged into LDS by the Tensor Data Mover (TDM) where
// the toolchain exposes it (s_wait_tensorcnt completes it), else by a plain
// cooperative copy. One v_wmma_f32_16x16x32_f16 computes votes for
// ipw = min(16/e, 32/d) capsule-input indices at once using a block-diagonal
// B fragment (K-slot s feeds only columns s*e .. s*e+e-1).
// Wave-uniform control flow around WMMA (EXEC must be all ones).
// ---------------------------------------------------------------------------
__global__ __launch_bounds__(128)
void capsule_conv_wmma(const float* __restrict__ in,           // [Cin, H, W]
                       const unsigned short* __restrict__ wq,  // f16 [n, I, d, e]
                       float* __restrict__ out,                // [n*e, H, W]
                       int W, int H, int k, int pad,
                       int I, int d, int e, int doSquash)
{
    extern __shared__ char smem[];
    const int L     = W * H;
    const int n     = blockIdx.y;
    const int ltile = blockIdx.x * 16;
    const int tid   = threadIdx.x;

    float*    votes = (float*)smem;                               // [16][I][e]
    _Float16* wl    = (_Float16*)(smem + (size_t)16 * I * e * 4); // [I][d][e]

    const _Float16* wf = (const _Float16*)wq + (size_t)n * I * d * e;
    const int wbytes = I * d * e * 2;   // multiple of 4 for all layers

    // Warm L2/WGP$ for the im2col gathers of this pixel tile.
    __builtin_prefetch(in + ltile + (tid & 15), 0, 3);

#if defined(__has_builtin) && __has_builtin(__builtin_amdgcn_tensor_load_to_lds)
    // ---- TDM: DMA the weight slice global -> LDS (ISA ch.7/8 D# layout) ----
    if (tid < 32) {  // wave 0 issues the tensor op (EXEC ignored by TDM)
        unsigned long long ga = (unsigned long long)(const void*)wf;
        unsigned int lds_addr = (unsigned int)(unsigned long long)wl;
        unsigned int nd = (unsigned int)(wbytes >> 2);   // DWORD elements
        v4u g0;
        g0[0] = 1u;                                       // count=1, user mode
        g0[1] = lds_addr;                                 // lds_addr [63:32]
        g0[2] = (unsigned int)ga;                         // global_addr lo
        g0[3] = (unsigned int)((ga >> 32) & 0x1FFFFFFull) // global_addr hi (57b)
                | (2u << 30);                             // type = 2 ("image")
        v8i g1;
        g1[0] = (int)(2u << 16);                          // mask=0, data_size=4B
        g1[1] = (int)((nd & 0xFFFFu) << 16);              // tensor_dim0 lo16
        g1[2] = (int)(((nd >> 16) & 0xFFFFu) | (1u << 16)); // dim0 hi16 | tensor_dim1=1
        g1[3] = (int)((nd & 0xFFFFu) << 16);              // tile_dim0 = nd
        g1[4] = 1;                                        // tile_dim1 = 1
        g1[5] = (int)nd;                                  // tensor_dim0_stride
        g1[6] = 0;
        g1[7] = 0;
        v4i z4 = {0, 0, 0, 0};
#if __clang_major__ >= 23
        v8i z8 = {0, 0, 0, 0, 0, 0, 0, 0};
        __builtin_amdgcn_tensor_load_to_lds(g0, g1, z4, z4, z8, 0);
#else
        __builtin_amdgcn_tensor_load_to_lds(g0, g1, z4, z4, 0);
#endif
        __builtin_amdgcn_s_wait_tensorcnt(0);
    }
#else
    for (int t = tid; t < I * d * e; t += 128) wl[t] = wf[t];
#endif
    __syncthreads();

    const int ipw  = (16 / e < 32 / d) ? (16 / e) : (32 / d);  // i's per WMMA
    const int lane = tid & 31;
    const int wave = tid >> 5;
    const int half = lane >> 4;     // K-range select per ISA A/B layout
    const int l16  = lane & 15;
    const int kk   = k * k;

    // -------- Phase 1: votes via WMMA, written straight to LDS --------------
    for (int ibase = wave * ipw; ibase < I; ibase += 4 * ipw) {
        // A fragment: M = pixel (0..15), K = slot*d + dd, im2col on the fly.
        v16h a;
        {
            const int pix = l16;
            const int gl  = ltile + pix;
            const int y0  = gl / W, x0 = gl % W;
#pragma unroll
            for (int j = 0; j < 16; ++j) {
                int v = j >> 1, p = j & 1;
                int K = ((v >= 4) ? 16 : 0) + (half ? 8 : 0) + ((v & 3) << 1) + p;
                int slot = K / d, dd = K % d;
                float val = 0.f;
                int i = ibase + slot;
                if (slot < ipw && i < I) {
                    int ci = i / kk, pp = i % kk;
                    int ky = pp / k, kx = pp % k;
                    int yy = y0 + ky - pad, xx = x0 + kx - pad;
                    if (yy >= 0 && yy < H && xx >= 0 && xx < W)
                        val = in[(size_t)(ci * d + dd) * L + yy * W + xx];
                }
                a[j] = (_Float16)val;
            }
        }
        // B fragment: block-diagonal — K-slot s feeds only columns s*e..s*e+e-1.
        const int col = l16;
        const int sn  = col / e, ee = col % e;
        v16h b;
#pragma unroll
        for (int j = 0; j < 16; ++j) {
            int v = j >> 1, p = j & 1;
            int K = ((v >= 4) ? 16 : 0) + (half ? 8 : 0) + ((v & 3) << 1) + p;
            int sk = K / d, dd = K % d;
            _Float16 val = (_Float16)0.f;
            int i = ibase + sn;
            if (sk == sn && sn < ipw && i < I)
                val = wl[(i * d + dd) * e + ee];
            b[j] = val;
        }
        v8f c = {};
        c = __builtin_amdgcn_wmma_f32_16x16x32_f16(
                /*neg_a=*/false, a, /*neg_b=*/false, b,
                /*c_mod=*/(short)0, c, /*reuse_a=*/false, /*reuse_b=*/false);
        // D layout: row M = r + 8*half, col = lane&15  ->  votes[pix][i][ee]
        {
            int i = ibase + sn;
            if (sn < ipw && i < I) {
#pragma unroll
                for (int r = 0; r < 8; ++r) {
                    int pixm = r + (half << 3);
                    votes[((size_t)pixm * I + i) * e + ee] = c[r];
                }
            }
        }
    }
    __syncthreads();

    // -------- Phase 2: fused routing (8 lanes per pixel, shfl width 8) ------
    const int pix = tid >> 3;
    const int ts  = tid & 7;
    const int gl2 = ltile + pix;

    float mu[16];
#pragma unroll
    for (int q = 0; q < 16; ++q) mu[q] = 0.f;
    for (int i = ts; i < I; i += 8) {
        const float* vp = &votes[((size_t)pix * I + i) * e];
#pragma unroll
        for (int q = 0; q < 16; ++q) if (q < e) mu[q] += vp[q];
    }
#pragma unroll
    for (int q = 0; q < 16; ++q) {
        if (q < e) {
            float s = mu[q];
            s += __shfl_xor(s, 1, 8);
            s += __shfl_xor(s, 2, 8);
            s += __shfl_xor(s, 4, 8);
            mu[q] = s / (float)I;
        }
    }

    for (int it = 0; it < 3; ++it) {
        float mumu = 0.f;
#pragma unroll
        for (int q = 0; q < 16; ++q) if (q < e) mumu += mu[q] * mu[q];

        // pass 1: max_i sim
        float mx = -3.0e38f;
        for (int i = ts; i < I; i += 8) {
            const float* vp = &votes[((size_t)pix * I + i) * e];
            float dot = 0.f, vv = 0.f;
#pragma unroll
            for (int q = 0; q < 16; ++q) if (q < e) {
                float v = vp[q]; dot += v * mu[q]; vv += v * v;
            }
            float sim = dot / (vv + mumu - dot);
            mx = fmaxf(mx, sim);
        }
        mx = fmaxf(mx, __shfl_xor(mx, 1, 8));
        mx = fmaxf(mx, __shfl_xor(mx, 2, 8));
        mx = fmaxf(mx, __shfl_xor(mx, 4, 8));

        // pass 2: softmax-weighted mean of votes
        float den = 0.f;
        float num[16];
#pragma unroll
        for (int q = 0; q < 16; ++q) num[q] = 0.f;
        for (int i = ts; i < I; i += 8) {
            const float* vp = &votes[((size_t)pix * I + i) * e];
            float dot = 0.f, vv = 0.f;
#pragma unroll
            for (int q = 0; q < 16; ++q) if (q < e) {
                float v = vp[q]; dot += v * mu[q]; vv += v * v;
            }
            float sim = dot / (vv + mumu - dot);
            float wgt = __expf(sim - mx);
            den += wgt;
#pragma unroll
            for (int q = 0; q < 16; ++q) if (q < e) num[q] += wgt * vp[q];
        }
        den += __shfl_xor(den, 1, 8);
        den += __shfl_xor(den, 2, 8);
        den += __shfl_xor(den, 4, 8);
#pragma unroll
        for (int q = 0; q < 16; ++q) {
            if (q < e) {
                float s = num[q];
                s += __shfl_xor(s, 1, 8);
                s += __shfl_xor(s, 2, 8);
                s += __shfl_xor(s, 4, 8);
                mu[q] = s / den;
            }
        }
    }

    if (ts == 0) {
        float scale = 1.f;
        if (doSquash) {
            float n2 = 0.f;
#pragma unroll
            for (int q = 0; q < 16; ++q) if (q < e) n2 += mu[q] * mu[q];
            scale = n2 / (1.f + n2) * rsqrtf(n2 + 1e-12f);
        }
        for (int q = 0; q < e; ++q)
            out[(size_t)(n * e + q) * L + gl2] = mu[q] * scale;
    }
}

// ---------------------------------------------------------------------------
// BatchNorm with batch statistics (B=1, stats over HxW); optional residual add.
// ---------------------------------------------------------------------------
__global__ __launch_bounds__(256)
void bn_kernel(const float* __restrict__ in, const float* __restrict__ g,
               const float* __restrict__ b, const float* __restrict__ base,
               float* __restrict__ out, int L)
{
    __shared__ float ssum[256];
    __shared__ float ssq[256];
    const int c = blockIdx.x;
    const float* p = in + (size_t)c * L;
    float s = 0.f, q = 0.f;
    for (int i = threadIdx.x; i < L; i += 256) { float v = p[i]; s += v; q += v * v; }
    ssum[threadIdx.x] = s; ssq[threadIdx.x] = q;
    __syncthreads();
    for (int st = 128; st > 0; st >>= 1) {
        if (threadIdx.x < st) {
            ssum[threadIdx.x] += ssum[threadIdx.x + st];
            ssq[threadIdx.x]  += ssq[threadIdx.x + st];
        }
        __syncthreads();
    }
    const float mean = ssum[0] / (float)L;
    const float var  = ssq[0] / (float)L - mean * mean;
    const float inv  = rsqrtf(var + 1e-5f) * g[c];
    const float bc   = b[c];
    float* o = out + (size_t)c * L;
    const float* bs = base ? base + (size_t)c * L : nullptr;
    for (int i = threadIdx.x; i < L; i += 256) {
        float v = (p[i] - mean) * inv + bc;
        o[i] = bs ? bs[i] + v : v;
    }
}

__global__ void add_kernel(const float* __restrict__ a, const float* __restrict__ b,
                           float* __restrict__ o, int n) {
    int i = blockIdx.x * 256 + threadIdx.x;
    if (i < n) o[i] = a[i] + b[i];
}

// (256,32,32) -> (64,64,64), r=2
__global__ void pixel_shuffle_kernel(const float* __restrict__ in, float* __restrict__ out) {
    int idx = blockIdx.x * 256 + threadIdx.x;
    if (idx < 64 * 64 * 64) {
        int co  = idx >> 12;
        int rem = idx & 4095;
        int Y = rem >> 6, X = rem & 63;
        int ic = co * 4 + ((Y & 1) << 1) + (X & 1);
        out[idx] = in[(size_t)ic * 1024 + (Y >> 1) * 32 + (X >> 1)];
    }
}

// ---------------------------------------------------------------------------
// Host orchestration
// ---------------------------------------------------------------------------
extern "C" void kernel_launch(void* const* d_in, const int* in_sizes, int n_in,
                              void* d_out, int out_size, void* d_ws, size_t ws_size,
                              hipStream_t stream) {
    (void)out_size; (void)ws_size;
    // ---- locate inputs (x = 3072 elems); handle sorted-key or insertion order
    int order[64]; int m = 0; int xi = 0;
    for (int i = 0; i < n_in && m < 64; ++i) {
        if (in_sizes[i] == 3072) { xi = i; continue; }
        order[m++] = i;
    }
    const float* x = (const float*)d_in[xi];
    auto F = [&](int j) { return (const float*)d_in[order[j]]; };

    const float *w1f = nullptr, *w8f = nullptr, *wupf = nullptr, *w10f = nullptr;
    const float *pw1[6], *pg1[6], *pb1[6], *pw2[6], *pg2[6], *pb2[6];
    if (m >= 40 && in_sizes[order[0]] == 64) {
        // jax-tree order: 6 x (b1,b2,g1,g2,w1,w2), then w1, w10, w8, w_up
        int p = 0;
        for (int r = 0; r < 6; ++r) {
            pb1[r] = F(p++); pb2[r] = F(p++); pg1[r] = F(p++);
            pg2[r] = F(p++); pw1[r] = F(p++); pw2[r] = F(p++);
        }
        w1f = F(p++); w10f = F(p++); w8f = F(p++); wupf = F(p++);
    } else {
        // insertion order: w1, 6 x (w1,g1,b1,w2,g2,b2), w8, w_up, w10
        int p = 0;
        w1f = F(p++);
        for (int r = 0; r < 6; ++r) {
            pw1[r] = F(p++); pg1[r] = F(p++); pb1[r] = F(p++);
            pw2[r] = F(p++); pg2[r] = F(p++); pb2[r] = F(p++);
        }
        w8f = F(p++); wupf = F(p++); w10f = F(p++);
    }

    // ---- workspace bump allocator (total ~5 MB)
    char* wsB = (char*)d_ws;
    size_t off = 0;
    auto alloc = [&](size_t bytes) -> void* {
        void* pp = wsB + off;
        off = (off + bytes + 255) & ~(size_t)255;
        return pp;
    };
    unsigned short* hw1  = (unsigned short*)alloc(15552 * 2);
    unsigned short* hw10 = (unsigned short*)alloc(15552 * 2);
    unsigned short* hw8  = (unsigned short*)alloc(36864 * 2);
    unsigned short* hwup = (unsigned short*)alloc((size_t)147456 * 2);
    unsigned short *hrw1[6], *hrw2[6];
    for (int r = 0; r < 6; ++r) {
        hrw1[r] = (unsigned short*)alloc(36864 * 2);
        hrw2[r] = (unsigned short*)alloc(36864 * 2);
    }
    float* b1buf = (float*)alloc((size_t)65536 * 4);
    float* hA    = (float*)alloc((size_t)65536 * 4);
    float* hB    = (float*)alloc((size_t)65536 * 4);
    float* T1    = (float*)alloc((size_t)65536 * 4);
    float* T2    = (float*)alloc((size_t)65536 * 4);
    float* UP    = (float*)alloc((size_t)262144 * 4);
    float* Yb    = (float*)alloc((size_t)262144 * 4);

    auto cvt = [&](const float* s, unsigned short* d, int nel) {
        f32_to_f16_kernel<<<(nel + 255) / 256, 256, 0, stream>>>(s, d, nel);
    };
    cvt(w1f, hw1, 15552);
    cvt(w10f, hw10, 15552);
    cvt(w8f, hw8, 36864);
    cvt(wupf, hwup, 147456);
    for (int r = 0; r < 6; ++r) { cvt(pw1[r], hrw1[r], 36864); cvt(pw2[r], hrw2[r], 36864); }

    auto conv = [&](const float* in, int Wd, const unsigned short* wq,
                    int nn, int I, int dd, int ee, int kk, int pp,
                    float* outp, int sq) {
        int L = Wd * Wd;
        size_t sh = (size_t)16 * I * ee * 4 + (size_t)I * dd * ee * 2;
        dim3 grid(L / 16, nn);
        capsule_conv_wmma<<<grid, dim3(128), sh, stream>>>(
            in, wq, outp, Wd, Wd, kk, pp, I, dd, ee, sq);
    };

    const int RB_IL[6] = {4, 8, 8, 16, 16, 8};
    const int RB_OL[6] = {8, 8, 16, 16, 8, 8};

    // b1 = capsule_conv(x, w1, 9, 4):  (3,32,32) -> (64,32,32)
    conv(x, 32, hw1, 16, 243, 1, 4, 9, 4, b1buf, 0);

    const float* hcur = b1buf;
    float* hnext = hA;
    for (int r = 0; r < 6; ++r) {
        const int il = RB_IL[r], ol = RB_OL[r];
        conv(hcur, 32, hrw1[r], 64 / ol, (64 / il) * 9, il, ol, 3, 1, T1, 0);
        bn_kernel<<<64, 256, 0, stream>>>(T1, pg1[r], pb1[r], nullptr, T2, 1024);
        conv(T2, 32, hrw2[r], 64 / ol, (64 / ol) * 9, ol, ol, 3, 1, T1, 0);
        bn_kernel<<<64, 256, 0, stream>>>(T1, pg2[r], pb2[r], hcur, hnext, 1024);
        hcur = hnext;
        hnext = (hcur == hA) ? hB : hA;
    }

    // b8 = capsule_conv(h, w8, 3, 1); S = b1 + b8
    conv(hcur, 32, hw8, 16, 72, 8, 4, 3, 1, T1, 0);
    add_kernel<<<256, 256, 0, stream>>>(b1buf, T1, T2, 65536);

    // up = capsule_conv(S, w_up, 3, 1): (64,32,32) -> (256,32,32); pixel shuffle
    conv(T2, 32, hwup, 64, 144, 4, 4, 3, 1, UP, 0);
    pixel_shuffle_kernel<<<1024, 256, 0, stream>>>(UP, Yb);

    // final: capsule_conv(y, w10, 9, 4, squash) -> (3,64,64) directly to d_out
    conv(Yb, 64, hw10, 3, 1296, 4, 1, 9, 4, (float*)d_out, 1);
}
